// CausalSelfAttention_24936580121098
// MI455X (gfx1250) — matrix operand order repaired and checked by reading, not compiled, not run on hardware
//
#include <hip/hip_runtime.h>
#include <hip/hip_bf16.h>

// ---------------------------------------------------------------------------
// Types
// ---------------------------------------------------------------------------
typedef __bf16 v16bf  __attribute__((ext_vector_type(16)));
typedef __bf16 bf16x8 __attribute__((ext_vector_type(8)));
typedef __bf16 bf16x4 __attribute__((ext_vector_type(4)));
typedef float  v8f    __attribute__((ext_vector_type(8)));
typedef int    v4i    __attribute__((vector_size(16)));   // gcc-style int4

#define BATCH        2
#define SEQ          2048
#define MODEL_DIM    2048
#define NUM_HEADS    32
#define NUM_KV_HEADS 8
#define HEAD_DIM     64

// ---------------------------------------------------------------------------
// gfx1250 async-LDS (validated: b128 variant takes
// (v4i addrspace(1)*, v4i addrspace(3)*, imm int offset, imm int cpol))
// ---------------------------------------------------------------------------
#if __has_builtin(__builtin_amdgcn_global_load_async_to_lds_b128) && \
    __has_builtin(__builtin_amdgcn_s_wait_asynccnt)
#define HAVE_ASYNC_LDS 1
#else
#define HAVE_ASYNC_LDS 0
#endif

#define AS1 __attribute__((address_space(1)))
#define AS3 __attribute__((address_space(3)))

// Generic LDS pointer low 32 bits == LDS offset (flat aperture rule,
// CDNA5 ISA).
__device__ __forceinline__ AS1 v4i* gbl_i4(const void* p) {
  return (AS1 v4i*)(unsigned long long)p;
}
__device__ __forceinline__ AS3 v4i* lds_i4(void* p) {
  return (AS3 v4i*)(unsigned int)(unsigned long long)p;
}

__device__ __forceinline__ void stage16B(const __bf16* g, __bf16* l) {
#if HAVE_ASYNC_LDS
  __builtin_amdgcn_global_load_async_to_lds_b128(gbl_i4(g), lds_i4(l), 0, 0);
#else
  *(bf16x8*)l = *(const bf16x8*)g;
#endif
}
// Wait until at most N async ops remain outstanding (N must be a literal).
__device__ __forceinline__ void stage_wait0() {
#if HAVE_ASYNC_LDS
  __builtin_amdgcn_s_wait_asynccnt(0);
#endif
}
__device__ __forceinline__ void stage_wait3() {
#if HAVE_ASYNC_LDS
  __builtin_amdgcn_s_wait_asynccnt(3);
#endif
}

// ---------------------------------------------------------------------------
// Helpers
// ---------------------------------------------------------------------------
__device__ __forceinline__ v16bf make_frag(bf16x8 lo, bf16x8 hi) {
  v16bf r;
#pragma unroll
  for (int i = 0; i < 8; ++i) { r[i] = lo[i]; r[8 + i] = hi[i]; }
  return r;
}

__device__ __forceinline__ v8f wmma_bf16(v16bf a, v16bf b, v8f c) {
  // D = A(16x32) * B(32x16) + C, fp32 accumulate
  return __builtin_amdgcn_wmma_f32_16x16x32_bf16(
      /*neg_a=*/false, a, /*neg_b=*/false, b,
      /*c_mod=*/(short)0, c, /*reuse_a=*/false, /*reuse_b=*/false);
}

// Per-wave ordering for private LDS regions (waves have divergent trip
// counts under the causal mask, so __syncthreads() is not usable).
__device__ __forceinline__ void wave_sync() {
  __builtin_amdgcn_fence(__ATOMIC_RELEASE, "wavefront");
  __builtin_amdgcn_wave_barrier();
  __builtin_amdgcn_fence(__ATOMIC_ACQUIRE, "wavefront");
}

// ---------------------------------------------------------------------------
// fp32 -> bf16 conversion (vectorized x4)
// ---------------------------------------------------------------------------
__global__ void cvt_f32_to_bf16(const float* __restrict__ in,
                                __bf16* __restrict__ out, size_t n4) {
  size_t i = (size_t)blockIdx.x * blockDim.x + threadIdx.x;
  if (i < n4) {
    float4 f = ((const float4*)in)[i];
    bf16x4 o;
    o[0] = (__bf16)f.x; o[1] = (__bf16)f.y; o[2] = (__bf16)f.z; o[3] = (__bf16)f.w;
    ((bf16x4*)out)[i] = o;
  }
}

// ---------------------------------------------------------------------------
// bf16 GEMM: C[M,N] = A[M,K] * B[K,N], row-major, fp32 accumulation.
// Block tile 64(M) x 128(N), 8 waves (4x2), each wave a 16x64 register tile
// (4 WMMAs per K-step). Double-buffered async-LDS staging: next K-step's
// tiles stream into the alternate buffer while WMMAs consume the current one
// (partial wait: s_wait_asynccnt 3 keeps the 3 new asyncs in flight).
// ---------------------------------------------------------------------------
__device__ __forceinline__ void gemm_stage(const __bf16* __restrict__ A,
                                           const __bf16* __restrict__ B,
                                           __bf16 (*sAb)[32],
                                           __bf16 (*sBb)[128], int tid,
                                           int blockM, int blockN, int K,
                                           int N, int k0) {
  // A tile 64x32: each thread one 16B chunk.
  const __bf16* ga = &A[(size_t)(blockM + (tid >> 2)) * K + k0 + (tid & 3) * 8];
  stage16B(ga, &sAb[tid >> 2][(tid & 3) * 8]);
  // B tile 32x128: each thread two 16B chunks.
  const __bf16* gb = &B[(size_t)(k0 + (tid >> 3)) * N + blockN + (tid & 7) * 16];
  __bf16*       lb = &sBb[tid >> 3][(tid & 7) * 16];
  stage16B(gb, lb);
  stage16B(gb + 8, lb + 8);
}

template <bool OUT_F32>
__global__ __launch_bounds__(256) void gemm_bf16_kernel(
    const __bf16* __restrict__ A, const __bf16* __restrict__ B,
    void* __restrict__ Cout, int M, int N, int K) {
  __shared__ __bf16 sA[2][64][32];    // 8 KB
  __shared__ __bf16 sB[2][32][128];   // 16 KB

  const int tid  = threadIdx.x;
  const int wave = tid >> 5;
  const int lane = tid & 31;
  const int wm   = wave >> 1;       // 0..3  (M sub-tile)
  const int wn   = wave & 1;        // 0..1  (N 64-wide sub-tile)
  const int hf   = lane >> 4;       // lane half
  const int ln   = lane & 15;
  const int kb   = hf * 8;          // unused; A-frag uses kbb too
  const int kbb  = hf * 16;         // B-frag K base

  const int blockM = blockIdx.x * 64;
  const int blockN = blockIdx.y * 128;

  v8f acc[4];
#pragma unroll
  for (int t = 0; t < 4; ++t)
    acc[t] = (v8f){0.f, 0.f, 0.f, 0.f, 0.f, 0.f, 0.f, 0.f};

  // Prologue: stage first K-step into buffer 0.
  gemm_stage(A, B, sA[0], sB[0], tid, blockM, blockN, K, N, 0);

  int cur = 0;
  for (int k0 = 0; k0 < K; k0 += 32) {
    const int knext = k0 + 32;
    if (knext < K) {
      // Stream next tiles into alternate buffer; only require the current
      // step's (oldest) 3 asyncs to have completed.
      gemm_stage(A, B, sA[cur ^ 1], sB[cur ^ 1], tid, blockM, blockN, K, N,
                 knext);
      stage_wait3();
    } else {
      stage_wait0();
    }
    __syncthreads();

    // Prefetch two K-steps ahead into cache while WMMAs run.
    if (k0 + 64 < K) {
      __builtin_prefetch(&A[(size_t)(blockM + (tid >> 2)) * K + k0 + 64], 0, 1);
      __builtin_prefetch(&B[(size_t)(k0 + 64 + (tid >> 3)) * N + blockN], 0, 1);
    }

    // A fragment (16x32): contiguous 16B chunks at K kbb and kbb+8.
    v16bf af = make_frag(*(const bf16x8*)&sA[cur][wm * 16 + ln][kbb],
                         *(const bf16x8*)&sA[cur][wm * 16 + ln][kbb + 8]);
#pragma unroll
    for (int t = 0; t < 4; ++t) {
      // B fragment (32x16): lane column = ln, element e -> k = kbb + e.
      v16bf bfv;
#pragma unroll
      for (int e = 0; e < 16; ++e)
        bfv[e] = sB[cur][kbb + e][wn * 64 + t * 16 + ln];
      acc[t] = wmma_bf16(af, bfv, acc[t]);
    }
    __syncthreads();  // all waves done reading buf 'cur' before it is reused
    cur ^= 1;
  }

  // Epilogue: C-frag element r -> (row = r + 8*hf, col = ln)
  const int colbase = blockN + wn * 64;
#pragma unroll
  for (int r = 0; r < 8; ++r) {
    const int row = blockM + wm * 16 + r + 8 * hf;
#pragma unroll
    for (int t = 0; t < 4; ++t) {
      const int col = colbase + t * 16 + ln;
      if (OUT_F32)
        ((float*)Cout)[(size_t)row * N + col] = acc[t][r];
      else
        ((__bf16*)Cout)[(size_t)row * N + col] = (__bf16)acc[t][r];
    }
  }
}

// ---------------------------------------------------------------------------
// RoPE (in-place, bf16), optional scale folded in (1/sqrt(HD) for Q).
// X layout: [b, s, h, d] with row stride heads*64. One thread per (row,h,i<32).
// ---------------------------------------------------------------------------
__global__ void rope_kernel(__bf16* __restrict__ X, int heads, float scale,
                            size_t total) {
  size_t idx = (size_t)blockIdx.x * blockDim.x + threadIdx.x;
  if (idx >= total) return;
  int i       = (int)(idx & 31);
  size_t rest = idx >> 5;            // (b*SEQ + s)*heads + h
  int h       = (int)(rest % heads);
  size_t row  = rest / heads;        // b*SEQ + s
  int s       = (int)(row % SEQ);

  float f   = __expf(-((float)(2 * i) * (1.0f / 64.0f)) * __logf(10000.0f));
  float ang = (float)s * f;
  float c   = __cosf(ang);
  float sn  = __sinf(ang);

  __bf16* p = X + row * (size_t)heads * HEAD_DIM + (size_t)h * HEAD_DIM;
  float q0 = (float)p[i];
  float q1 = (float)p[i + 32];
  p[i]      = (__bf16)((q0 * c - q1 * sn) * scale);
  p[i + 32] = (__bf16)((q1 * c + q0 * sn) * scale);
}

// ---------------------------------------------------------------------------
// Flash attention (causal, online softmax). One wave per 16-query tile.
// Q layout [b,s,h,d] (stride 2048); K/V layout [b,s,hkv,d] (stride 512).
// Scale 1/sqrt(HD) already folded into Q by rope_kernel.
// V staging is issued asynchronously and overlapped with the QK^T WMMAs.
// ---------------------------------------------------------------------------
__global__ __launch_bounds__(256) void flash_attn_kernel(
    const __bf16* __restrict__ Q, const __bf16* __restrict__ K,
    const __bf16* __restrict__ V, __bf16* __restrict__ Ctx) {
  __shared__ __bf16 sP[8][16][32];   // 8 KB  : per-wave P tile (probs)
  __shared__ __bf16 sV[8][32][64];   // 32 KB : per-wave V chunk

  const int wave = threadIdx.x >> 5;
  const int lane = threadIdx.x & 31;
  const int hf   = lane >> 4;
  const int ln   = lane & 15;
  const int kb   = hf * 8;
  const int kbb  = hf * 16;

  const int bh  = blockIdx.x;        // 0 .. B*NUM_HEADS-1
  const int b   = bh >> 5;
  const int h   = bh & 31;
  const int hkv = h >> 2;            // 4 query heads per kv head
  const int s0  = (blockIdx.y * 8 + wave) * 16;

  const int qstride  = NUM_HEADS * HEAD_DIM;     // 2048
  const int kvstride = NUM_KV_HEADS * HEAD_DIM;  // 512

  // Q A-fragments (d0 = 0 and d0 = 32), loop-invariant.
  const __bf16* qrow =
      Q + ((size_t)(b * SEQ + s0 + ln) * qstride + h * HEAD_DIM);
  v16bf aq0 = make_frag(*(const bf16x8*)(qrow + kbb),
                        *(const bf16x8*)(qrow + kbb + 8));
  v16bf aq1 = make_frag(*(const bf16x8*)(qrow + 32 + kbb),
                        *(const bf16x8*)(qrow + 40 + kbb));

  v8f o[4];
#pragma unroll
  for (int t = 0; t < 4; ++t) o[t] = (v8f){0.f,0.f,0.f,0.f,0.f,0.f,0.f,0.f};
  float m[8], l[8];
#pragma unroll
  for (int r = 0; r < 8; ++r) { m[r] = -3.0e38f; l[r] = 0.0f; }

  const int nchunk = (s0 + 16 + 31) >> 5;  // 32-key chunks covering causal span
  for (int c = 0; c < nchunk; ++c) {
    const int j0 = c << 5;

    // Issue async V staging: lane L copies 64B (d 0..31) of key row j0+L into LDS.
    // No wait yet -- the QK^T WMMAs below don't touch sV.
    const __bf16* vrow =
        V + ((size_t)(b * SEQ + j0 + lane) * kvstride + hkv * HEAD_DIM);
    __bf16* lv = &sV[wave][lane][0];
    stage16B(vrow,      lv);
    stage16B(vrow + 8,  lv + 8);
    stage16B(vrow + 16, lv + 16);
    stage16B(vrow + 24, lv + 24);

    // Scores: S = Q(16x64) @ K^T(64x32) via 4 WMMA ops (global K loads land
    // directly in B-frag shape: lane column = key, element e -> d = d0+kbb+e).
    v8f sc0 = (v8f){0.f,0.f,0.f,0.f,0.f,0.f,0.f,0.f};
    v8f sc1 = sc0;
    const __bf16* krow0 =
        K + ((size_t)(b * SEQ + j0 + ln) * kvstride + hkv * HEAD_DIM);
    {
      v16bf bk = make_frag(*(const bf16x8*)(krow0 + kbb),
                           *(const bf16x8*)(krow0 + kbb + 8));
      sc0 = wmma_bf16(aq0, bk, sc0);
      bk = make_frag(*(const bf16x8*)(krow0 + 32 + kbb),
                     *(const bf16x8*)(krow0 + 32 + kbb + 8));
      sc0 = wmma_bf16(aq1, bk, sc0);

      const __bf16* krow1 = krow0 + (size_t)16 * kvstride;
      bk = make_frag(*(const bf16x8*)(krow1 + kbb),
                     *(const bf16x8*)(krow1 + kbb + 8));
      sc1 = wmma_bf16(aq0, bk, sc1);
      bk = make_frag(*(const bf16x8*)(krow1 + 32 + kbb),
                     *(const bf16x8*)(krow1 + 32 + kbb + 8));
      sc1 = wmma_bf16(aq1, bk, sc1);
    }

    // Prefetch next chunk's K/V rows while softmax + PV run.
    if (c + 1 < nchunk) {
      __builtin_prefetch(vrow + (size_t)32 * kvstride, 0, 1);
      __builtin_prefetch(krow0 + (size_t)32 * kvstride, 0, 1);
    }

    // V staging must have landed before the PV gathers below.
    stage_wait0();
    wave_sync();

    // Causal mask + online softmax (row r of C-frag = query s0 + r + 8*hf).
#pragma unroll
    for (int r = 0; r < 8; ++r) {
      const int qr = s0 + 8 * hf + r;
      float v0 = (j0 + ln      > qr) ? -3.0e38f : sc0[r];
      float v1 = (j0 + 16 + ln > qr) ? -3.0e38f : sc1[r];

      float mx = fmaxf(v0, v1);
#pragma unroll
      for (int off = 8; off >= 1; off >>= 1)
        mx = fmaxf(mx, __shfl_xor(mx, off));
      float mnew  = fmaxf(m[r], mx);
      float alpha = __expf(m[r] - mnew);
      m[r] = mnew;

      float p0 = __expf(v0 - mnew);
      float p1 = __expf(v1 - mnew);
      float ps = p0 + p1;
#pragma unroll
      for (int off = 8; off >= 1; off >>= 1)
        ps += __shfl_xor(ps, off);
      l[r] = l[r] * alpha + ps;

#pragma unroll
      for (int t = 0; t < 4; ++t) o[t][r] *= alpha;

      const int prow = r + 8 * hf;
      sP[wave][prow][ln]      = (__bf16)p0;
      sP[wave][prow][16 + ln] = (__bf16)p1;
    }
    wave_sync();

    // O += P(16x32) @ V(32x64): re-shape P C-frag -> A-frag through LDS.
    const __bf16* prow = &sP[wave][ln][0];
    v16bf ap = make_frag(*(const bf16x8*)(prow + kbb),
                         *(const bf16x8*)(prow + kbb + 8));
#pragma unroll
    for (int t = 0; t < 4; ++t) {
      v16bf bv;
#pragma unroll
      for (int e = 0; e < 16; ++e) bv[e] = sV[wave][kbb + e][t * 16 + ln];
      o[t] = wmma_bf16(ap, bv, o[t]);
    }
    wave_sync();  // protect sP/sV from next iteration's writes
  }

  // Epilogue: normalize and write context [b,s,h,d] bf16.
#pragma unroll
  for (int r = 0; r < 8; ++r) {
    const int row = s0 + 8 * hf + r;
    const float inv = 1.0f / l[r];
    __bf16* crow = Ctx + ((size_t)(b * SEQ + row) * qstride + h * HEAD_DIM);
#pragma unroll
    for (int t = 0; t < 4; ++t)
      crow[t * 16 + ln] = (__bf16)(o[t][r] * inv);
  }
}

// ---------------------------------------------------------------------------
// Host-side launch
// ---------------------------------------------------------------------------
extern "C" void kernel_launch(void* const* d_in, const int* in_sizes, int n_in,
                              void* d_out, int out_size, void* d_ws,
                              size_t ws_size, hipStream_t stream) {
  (void)in_sizes; (void)n_in; (void)out_size; (void)ws_size;
  const float* x  = (const float*)d_in[0];
  const float* Wq = (const float*)d_in[1];
  const float* Wk = (const float*)d_in[2];
  const float* Wv = (const float*)d_in[3];
  const float* Wo = (const float*)d_in[4];
  float* out = (float*)d_out;

  const size_t M   = (size_t)BATCH * SEQ;                           // 4096
  const size_t nX  = M * MODEL_DIM;                                 // 8388608
  const size_t nWq = (size_t)MODEL_DIM * NUM_HEADS * HEAD_DIM;      // 4194304
  const size_t nWk = (size_t)MODEL_DIM * NUM_KV_HEADS * HEAD_DIM;   // 1048576
  const size_t nQ  = M * NUM_HEADS * HEAD_DIM;                      // 8388608
  const size_t nKV = M * NUM_KV_HEADS * HEAD_DIM;                   // 2097152

  __bf16* ws  = (__bf16*)d_ws;
  __bf16* xb  = ws;                 ws += nX;
  __bf16* wqb = ws;                 ws += nWq;
  __bf16* wkb = ws;                 ws += nWk;
  __bf16* wvb = ws;                 ws += nWk;
  __bf16* wob = ws;                 ws += nWq;
  __bf16* qb  = ws;                 ws += nQ;
  __bf16* kbuf= ws;                 ws += nKV;
  __bf16* vbuf= ws;                 ws += nKV;
  __bf16* ctx = ws;                 ws += nQ;

  // 1) fp32 -> bf16 conversions
  auto cvt = [&](const float* src, __bf16* dst, size_t n) {
    size_t n4 = n / 4;
    cvt_f32_to_bf16<<<dim3((unsigned)((n4 + 255) / 256)), dim3(256), 0,
                      stream>>>(src, dst, n4);
  };
  cvt(x,  xb,  nX);
  cvt(Wq, wqb, nWq);
  cvt(Wk, wkb, nWk);
  cvt(Wv, wvb, nWk);
  cvt(Wo, wob, nWq);

  // 2) QKV projections (bf16 WMMA GEMM, bf16 out)
  gemm_bf16_kernel<false><<<dim3(M / 64, (NUM_HEADS * HEAD_DIM) / 128), 256, 0,
                            stream>>>(xb, wqb, qb, (int)M,
                                      NUM_HEADS * HEAD_DIM, MODEL_DIM);
  gemm_bf16_kernel<false><<<dim3(M / 64, (NUM_KV_HEADS * HEAD_DIM) / 128), 256,
                            0, stream>>>(xb, wkb, kbuf, (int)M,
                                         NUM_KV_HEADS * HEAD_DIM, MODEL_DIM);
  gemm_bf16_kernel<false><<<dim3(M / 64, (NUM_KV_HEADS * HEAD_DIM) / 128), 256,
                            0, stream>>>(xb, wvb, vbuf, (int)M,
                                         NUM_KV_HEADS * HEAD_DIM, MODEL_DIM);

  // 3) RoPE (scale 1/sqrt(64)=0.125 folded into Q)
  {
    size_t totQ = M * NUM_HEADS * 32;
    rope_kernel<<<dim3((unsigned)((totQ + 255) / 256)), dim3(256), 0, stream>>>(
        qb, NUM_HEADS, 0.125f, totQ);
    size_t totK = M * NUM_KV_HEADS * 32;
    rope_kernel<<<dim3((unsigned)((totK + 255) / 256)), dim3(256), 0, stream>>>(
        kbuf, NUM_KV_HEADS, 1.0f, totK);
  }

  // 4) Causal flash attention -> ctx (bf16)
  flash_attn_kernel<<<dim3(BATCH * NUM_HEADS, SEQ / (16 * 8)), 256, 0,
                      stream>>>(qb, kbuf, vbuf, ctx);

  // 5) Output projection (fp32 out)
  gemm_bf16_kernel<true><<<dim3(M / 64, MODEL_DIM / 32 / 4), 256, 0, stream>>>(
      ctx, wob, (void*)out, (int)M, MODEL_DIM, NUM_HEADS * HEAD_DIM);
}
